// CLSTMEventNet_8452495638670
// MI455X (gfx1250) — matrix operand order repaired and checked by reading, not compiled, not run on hardware
//
#include <hip/hip_runtime.h>
#include <hip/hip_bf16.h>

typedef __attribute__((ext_vector_type(16))) __bf16        v16bf;
typedef __attribute__((ext_vector_type(8)))  float         v8f;
typedef __attribute__((ext_vector_type(4)))  unsigned int  u32x4;

constexpr int kB    = 8;
constexpr int kCin  = 32;
constexpr int kT    = 8;
constexpr int kH    = 48;
constexpr int kW    = 48;
constexpr int kHW   = kH * kW;          // 2304
constexpr int kNPIX = kB * kHW;         // 18432
constexpr int kHC   = 64;
constexpr int kCLS  = 10;
constexpr int kFEAT = kHC * kHW;        // 147456

// ---------------------------------------------------------------- helpers
__device__ __forceinline__ unsigned short f2bf(float f) {
    unsigned int u = __float_as_uint(f);
    unsigned int r = u + 0x7FFFu + ((u >> 16) & 1u);   // round-to-nearest-even
    return (unsigned short)(r >> 16);
}
__device__ __forceinline__ float bf2f(unsigned short h) {
    return __uint_as_float(((unsigned int)h) << 16);
}
__device__ __forceinline__ float sigm(float x) {
    return 1.0f / (1.0f + __expf(-x));
}

union Frag { u32x4 u[2]; v16bf v; };

__device__ __forceinline__ v8f wmma_bf16(const v16bf& a, const v16bf& b, const v8f& c) {
    // D = A(16x32 bf16) * B(32x16 bf16) + C(16x16 f32)
    return __builtin_amdgcn_wmma_f32_16x16x32_bf16(false, a, false, b, (short)0, c, false, false);
}

__device__ __forceinline__ Frag loadB(const unsigned short* p, bool ok) {
    Frag f;
    if (ok) {
        const u32x4* q = reinterpret_cast<const u32x4*>(p);
        f.u[0] = q[0]; f.u[1] = q[1];
    } else {
        u32x4 z = {0u, 0u, 0u, 0u};
        f.u[0] = z; f.u[1] = z;
    }
    return f;
}

// ------------------------------------------------- x: NCTHW f32 -> per-t NHWC bf16
__global__ void xconv_kernel(const float* __restrict__ x, unsigned short* __restrict__ out, int total) {
    int e = blockIdx.x * 256 + threadIdx.x;
    if (e >= total) return;
    int c = e & 31;
    int p = (e >> 5) % kNPIX;
    int t = (e >> 5) / kNPIX;
    int b = p / kHW;
    int yx = p % kHW;
    size_t idx = (((size_t)b * kCin + c) * kT + t) * kHW + yx;   // [B,Cin,T,H,W]
    out[e] = f2bf(x[idx]);
}

// ------------------------------------------------- weights -> A-fragment register image
// Layout read by cell kernel: aswz[((mt*nkc + kc)*32 + lane)*16 + j]
// A 16x32 layout (contiguous k order, matches B loads): m=lane&15, g=lane>>4, K = kc*32 + 16*g + j
__global__ void wprep_kernel(const float* __restrict__ Wx, const float* __restrict__ Wh,
                             int Ca, int Cb, int nkc, unsigned short* __restrict__ out, int total) {
    int e = blockIdx.x * 256 + threadIdx.x;
    if (e >= total) return;
    int j    = e & 15;
    int lane = (e >> 4) & 31;
    int rest = e >> 9;                // mt*nkc + kc
    int kc   = rest % nkc;
    int mt   = rest / nkc;
    int m    = lane & 15;
    int g    = lane >> 4;
    int co   = mt * 16 + m;
    int kk   = g * 16 + j;
    int K    = kc * 32 + kk;
    int Ctot = Ca + Cb;
    int tap  = K / Ctot;              // K ordering: (tap, channel)
    int ci   = K % Ctot;
    int ky   = tap / 3, kx = tap % 3;
    float v;
    if (ci < Ca) v = Wx[((size_t)(co * Ca + ci) * 3 + ky) * 3 + kx];          // OIHW
    else         v = Wh[((size_t)(co * Cb + (ci - Ca)) * 3 + ky) * 3 + kx];
    out[e] = f2bf(v);
}

// ------------------------------------------------- fused gates-GEMM + LSTM cell
// gates[256 x 32pix] = Wcat(256 x K) * im2col(K x 32pix), K = (Ca+Cb)*9
template <int CA, int CB>
__global__ __launch_bounds__(256) void cell_kernel(const unsigned short* __restrict__ inA,
                                                   const unsigned short* __restrict__ inB,
                                                   const unsigned short* __restrict__ aswz,
                                                   const float* __restrict__ bias,
                                                   float* __restrict__ cbuf,
                                                   unsigned short* __restrict__ hout) {
    constexpr int CTOT = CA + CB;
    constexpr int NSA  = CA / 32;       // 32-channel slices from tensor A per tap
    constexpr int NSB  = CB / 32;       // 32-channel slices from tensor B per tap
    constexpr int NKC  = (CTOT / 32) * 9;

    __shared__ float lds[128 * 33];     // half of the 256-row gate tile, padded stride

    const int tid  = threadIdx.x;
    const int lane = tid & 31;
    const int wave = tid >> 5;
    const int g    = lane >> 4;         // lane group (0: lanes 0-15, 1: lanes 16-31)
    const int ln   = lane & 15;
    const int g16  = g * 16;            // this lane-group's 16 contiguous channels
    const int pixbase = blockIdx.x * 32;

    const v8f zf = {0.f, 0.f, 0.f, 0.f, 0.f, 0.f, 0.f, 0.f};
    v8f acc[4] = {zf, zf, zf, zf};      // [mt2][nt]

    // two pixels per lane (one per N tile)
    const int p0 = pixbase + ln;
    const int p1 = pixbase + 16 + ln;
    const int b0 = p0 / kHW, y0 = (p0 % kHW) / kW, x0 = p0 % kW;
    const int b1 = p1 / kHW, y1 = (p1 % kHW) / kW, x1 = p1 % kW;

    // running A-fragment pointers (pre-swizzled register image, 64 u32x4 per chunk)
    const u32x4* abase = reinterpret_cast<const u32x4*>(aswz);
    const u32x4* a0p = abase + (size_t)(wave * 2)     * NKC * 64 + lane * 2;
    const u32x4* a1p = abase + (size_t)(wave * 2 + 1) * NKC * 64 + lane * 2;

    #pragma unroll 1
    for (int tap = 0; tap < 9; ++tap) {
        const int ky = tap / 3 - 1;
        const int kx = tap % 3 - 1;
        const int yy0 = y0 + ky, xx0 = x0 + kx;
        const int yy1 = y1 + ky, xx1 = x1 + kx;
        const bool ok0 = ((unsigned)yy0 < (unsigned)kH) & ((unsigned)xx0 < (unsigned)kW);
        const bool ok1 = ((unsigned)yy1 < (unsigned)kH) & ((unsigned)xx1 < (unsigned)kW);
        const size_t q0 = (size_t)((b0 * kH + yy0) * kW + xx0);   // not deref'd unless ok
        const size_t q1 = (size_t)((b1 * kH + yy1) * kW + xx1);
        const unsigned short* sA0 = inA + q0 * CA + g16;
        const unsigned short* sA1 = inA + q1 * CA + g16;
        const unsigned short* sB0 = inB + q0 * CB + g16;
        const unsigned short* sB1 = inB + q1 * CB + g16;

        #pragma unroll
        for (int s = 0; s < NSA; ++s) {
            Frag fb0 = loadB(sA0 + s * 32, ok0);
            Frag fb1 = loadB(sA1 + s * 32, ok1);
            Frag fa0, fa1;
            fa0.u[0] = a0p[0]; fa0.u[1] = a0p[1];
            fa1.u[0] = a1p[0]; fa1.u[1] = a1p[1];
            a0p += 64; a1p += 64;
            acc[0] = wmma_bf16(fa0.v, fb0.v, acc[0]);
            acc[1] = wmma_bf16(fa0.v, fb1.v, acc[1]);
            acc[2] = wmma_bf16(fa1.v, fb0.v, acc[2]);
            acc[3] = wmma_bf16(fa1.v, fb1.v, acc[3]);
        }
        #pragma unroll
        for (int s = 0; s < NSB; ++s) {
            Frag fb0 = loadB(sB0 + s * 32, ok0);
            Frag fb1 = loadB(sB1 + s * 32, ok1);
            Frag fa0, fa1;
            fa0.u[0] = a0p[0]; fa0.u[1] = a0p[1];
            fa1.u[0] = a1p[0]; fa1.u[1] = a1p[1];
            a0p += 64; a1p += 64;
            acc[0] = wmma_bf16(fa0.v, fb0.v, acc[0]);
            acc[1] = wmma_bf16(fa0.v, fb1.v, acc[1]);
            acc[2] = wmma_bf16(fa1.v, fb0.v, acc[2]);
            acc[3] = wmma_bf16(fa1.v, fb1.v, acc[3]);
        }
    }

    // ------- epilogue: combine 4 gates across waves via LDS (two phases)
    auto writeAccs = [&](int rowbase) {
        #pragma unroll
        for (int q = 0; q < 4; ++q) {
            const int mt2 = q >> 1, nt = q & 1;
            const int col = nt * 16 + ln;
            const int rb  = rowbase + mt2 * 16 + g * 8;   // C/D layout: lanes16-31 hold M=r+8
            #pragma unroll
            for (int r = 0; r < 8; ++r) lds[(rb + r) * 33 + col] = acc[q][r];
        }
    };

    const int e0  = tid * 8;            // 2048 = 64 hc x 32 pix, 8 per thread
    const int hc  = e0 >> 5;
    const int px0 = e0 & 31;

    float si[8], tc[8];

    if (wave < 4) writeAccs(wave * 32);             // gate rows 0..127 (i, f)
    __syncthreads();
    #pragma unroll
    for (int k = 0; k < 8; ++k) {
        const int px = px0 + k;
        const float xi = lds[hc * 33 + px]        + bias[hc];
        const float xf = lds[(64 + hc) * 33 + px] + bias[64 + hc];
        const float cold = cbuf[(size_t)(pixbase + px) * kHC + hc];
        si[k] = sigm(xi);
        tc[k] = sigm(xf) * cold;                    // f*c (peepholes are zero)
    }
    __syncthreads();
    if (wave >= 4) writeAccs((wave - 4) * 32);      // gate rows 128..255 (g, o)
    __syncthreads();
    #pragma unroll
    for (int k = 0; k < 8; ++k) {
        const int px = px0 + k;
        const float xg = lds[hc * 33 + px]        + bias[128 + hc];
        const float xo = lds[(64 + hc) * 33 + px] + bias[192 + hc];
        const float cc = tc[k] + si[k] * tanhf(xg);
        const float h  = sigm(xo) * tanhf(cc);
        const size_t idx = (size_t)(pixbase + px) * kHC + hc;
        cbuf[idx] = cc;
        hout[idx] = f2bf(h);
    }
}

// ------------------------------------------------- FC: logits[b][cls]
__global__ __launch_bounds__(256) void fc_kernel(const unsigned short* __restrict__ h2,
                                                 const float* __restrict__ Wfc,
                                                 const float* __restrict__ bfc,
                                                 float* __restrict__ logits) {
    const int b   = blockIdx.x / kCLS;
    const int cls = blockIdx.x % kCLS;
    const float* wrow = Wfc + (size_t)cls * kFEAT;
    float s = 0.f;
    for (int f = threadIdx.x; f < kFEAT; f += 256) {
        const int ch  = f / kHW;          // reference flattens NCHW
        const int rem = f % kHW;
        s += bf2f(h2[((size_t)b * kHW + rem) * kHC + ch]) * wrow[f];
    }
    __shared__ float red[256];
    red[threadIdx.x] = s;
    __syncthreads();
    for (int st = 128; st > 0; st >>= 1) {
        if (threadIdx.x < st) red[threadIdx.x] += red[threadIdx.x + st];
        __syncthreads();
    }
    if (threadIdx.x == 0) logits[b * kCLS + cls] = red[0] + bfc[cls];
}

__global__ void softmax_kernel(const float* __restrict__ logits, float* __restrict__ out) {
    const int b = threadIdx.x;
    if (b < kB) {
        float m = -1e30f;
        for (int c = 0; c < kCLS; ++c) m = fmaxf(m, logits[b * kCLS + c]);
        float e[kCLS], s = 0.f;
        for (int c = 0; c < kCLS; ++c) { e[c] = __expf(logits[b * kCLS + c] - m); s += e[c]; }
        for (int c = 0; c < kCLS; ++c) out[b * kCLS + c] = e[c] / s;
    }
}

// ------------------------------------------------- host orchestration
extern "C" void kernel_launch(void* const* d_in, const int* in_sizes, int n_in,
                              void* d_out, int out_size, void* d_ws, size_t ws_size,
                              hipStream_t stream) {
    (void)in_sizes; (void)n_in; (void)out_size; (void)ws_size;

    const float* x   = (const float*)d_in[0];
    const float* Wx0 = (const float*)d_in[1];
    const float* bx0 = (const float*)d_in[2];
    const float* Wh0 = (const float*)d_in[3];
    const float* Wx1 = (const float*)d_in[4];
    const float* bx1 = (const float*)d_in[5];
    const float* Wh1 = (const float*)d_in[6];
    const float* Wx2 = (const float*)d_in[7];
    const float* bx2 = (const float*)d_in[8];
    const float* Wh2 = (const float*)d_in[9];
    /* d_in[10] = peep (zeros, algebraically a no-op) */
    const float* Wfc = (const float*)d_in[11];
    const float* bfc = (const float*)d_in[12];

    // workspace carve-up (256B aligned)
    char* ws = (char*)d_ws;
    size_t off = 0;
    auto carve = [&](size_t bytes) { char* p = ws + off; off = (off + bytes + 255) & ~(size_t)255; return p; };

    const size_t xtBytes = (size_t)kT * kNPIX * kCin * 2;    // bf16 NHWC per t
    const size_t hBytes  = (size_t)kNPIX * kHC * 2;          // bf16 NHWC
    const size_t cBytes  = (size_t)kNPIX * kHC * 4;          // f32  NHWC
    const int nkc0 = (kCin + kHC) * 9 / 32;                  // 27
    const int nkc1 = (kHC + kHC) * 9 / 32;                   // 36
    const size_t a0Bytes = (size_t)16 * nkc0 * 512 * 2;
    const size_t a1Bytes = (size_t)16 * nkc1 * 512 * 2;

    unsigned short* xt_all = (unsigned short*)carve(xtBytes);
    unsigned short* hbuf[3][2];
    for (int l = 0; l < 3; ++l)
        for (int v = 0; v < 2; ++v) hbuf[l][v] = (unsigned short*)carve(hBytes);
    float* cb[3];
    for (int l = 0; l < 3; ++l) cb[l] = (float*)carve(cBytes);
    unsigned short* aswz0 = (unsigned short*)carve(a0Bytes);
    unsigned short* aswz1 = (unsigned short*)carve(a1Bytes);
    unsigned short* aswz2 = (unsigned short*)carve(a1Bytes);
    float* logits = (float*)carve(kB * kCLS * sizeof(float));

    // zero initial state: c=0, and the h buffer read at t=0 (index (0+1)&1 == 1)
    for (int l = 0; l < 3; ++l) {
        hipMemsetAsync(cb[l], 0, cBytes, stream);
        hipMemsetAsync(hbuf[l][1], 0, hBytes, stream);
    }

    // input conversion + weight pre-swizzle
    {
        const int total = kT * kNPIX * kCin;
        xconv_kernel<<<total / 256, 256, 0, stream>>>(x, xt_all, total);
    }
    {
        int t0 = 16 * nkc0 * 512;
        wprep_kernel<<<(t0 + 255) / 256, 256, 0, stream>>>(Wx0, Wh0, kCin, kHC, nkc0, aswz0, t0);
        int t1 = 16 * nkc1 * 512;
        wprep_kernel<<<(t1 + 255) / 256, 256, 0, stream>>>(Wx1, Wh1, kHC, kHC, nkc1, aswz1, t1);
        wprep_kernel<<<(t1 + 255) / 256, 256, 0, stream>>>(Wx2, Wh2, kHC, kHC, nkc1, aswz2, t1);
    }

    const int nblk = kNPIX / 32;   // 576
    for (int t = 0; t < kT; ++t) {
        const int rd = (t + 1) & 1, wr = t & 1;   // h(t) lives in buffer t&1
        const unsigned short* xt = xt_all + (size_t)t * kNPIX * kCin;

        cell_kernel<kCin, kHC><<<nblk, 256, 0, stream>>>(
            xt, hbuf[0][rd], aswz0, bx0, cb[0], hbuf[0][wr]);
        cell_kernel<kHC, kHC><<<nblk, 256, 0, stream>>>(
            hbuf[0][wr], hbuf[1][rd], aswz1, bx1, cb[1], hbuf[1][wr]);
        cell_kernel<kHC, kHC><<<nblk, 256, 0, stream>>>(
            hbuf[1][wr], hbuf[2][rd], aswz2, bx2, cb[2], hbuf[2][wr]);
    }

    fc_kernel<<<kB * kCLS, 256, 0, stream>>>(hbuf[2][(kT - 1) & 1], Wfc, bfc, logits);
    softmax_kernel<<<1, 32, 0, stream>>>(logits, (float*)d_out);
}
